// HungarianMatcher_Curve_61624190763776
// MI455X (gfx1250) — compile-verified, matrix operand order checked by reading, not run on hardware
//
#include <hip/hip_runtime.h>
#include <hip/hip_bf16.h>

// Problem constants (match reference)
constexpr int Qn   = 1024;
constexpr int Tn   = 256;
constexpr int Nn   = 100;
constexpr int K3   = 300;   // 3*N  (GEMM K)
constexpr int NKT  = 10;    // K tiles (K padded 300 -> 320)
constexpr int V2N  = 200;   // 2*N variant rows
constexpr int NVT  = 13;    // ceil(200/16) v tiles
constexpr int WAVES = 4;    // 128-thread block = 4 wave32
constexpr int AEXT = 640;   // doubled-curve array, zero-padded: max idx = 300+319 = 619
constexpr float EPSF = 1e-6f;

typedef __attribute__((ext_vector_type(16))) __bf16 v16bf;
typedef __attribute__((ext_vector_type(8)))  float  v8f;

// WMMA bf16 fragment K index for element i, half-wave hi (wave32):
// lanes 0-15 hold K {0..7,16..23}, lanes 16-31 hold K {8..15,24..31}
__device__ __forceinline__ int kmap(int i, int hi) {
    return i + (hi ? 8 : 0) + (i >= 8 ? 8 : 0);
}

__global__ __launch_bounds__(128)
void matcher_wmma_kernel(const float* __restrict__ preds,   // Q*300
                         const float* __restrict__ vlog,    // Q*2
                         const float* __restrict__ tlog,    // Q*4
                         const float* __restrict__ clog,    // Q*2
                         const float* __restrict__ tgt,     // T*300
                         const int*   __restrict__ labels,  // T
                         const int*   __restrict__ iscl,    // T
                         const float* __restrict__ wgt,     // T
                         float* __restrict__ outC,          // Q*T
                         float* __restrict__ outI)          // Q*T (ids as float)
{
    // B staged in WMMA-fragment-native order: Bfrag[kt][lane][elem] (32B/lane/frag)
    __shared__ __align__(16) __bf16 Bfrag[NKT * 32 * 16];
    __shared__ __align__(16) __bf16 tgtd[AEXT];   // doubled cyclic curve, zero tail
    __shared__ __align__(16) __bf16 revd[AEXT];   // doubled reversed curve, zero tail
    __shared__ float  p2s[16];
    __shared__ float  v2s;
    __shared__ float  wmin[WAVES][16];
    __shared__ int    widd[WAVES][16];
    __shared__ float  wo0[WAVES][16];
    __shared__ float  wo1[WAVES][16];

    const int tid  = threadIdx.x;
    const int lane = tid & 31;
    const int wave = tid >> 5;
    const int qt   = blockIdx.x;          // 0..63 (Q/16)
    const int t    = blockIdx.y;          // 0..255
    const int qbase = qt * 16;

    if (tid == 0) v2s = 0.0f;
    __syncthreads();

    // --- Stage B in fragment order: elem (kt,ln,i) <- preds bf16, zero-pad K ---
    for (int idx = tid; idx < NKT * 32 * 16; idx += 128) {
        int kt  = idx >> 9;               // /512
        int rem = idx & 511;
        int ln  = rem >> 4;
        int i   = rem & 15;
        int k   = kt * 32 + kmap(i, ln >> 4);
        int qq  = ln & 15;
        float v = (k < K3) ? preds[(qbase + qq) * K3 + k] : 0.0f;
        Bfrag[idx] = (__bf16)v;
    }
    // --- Stage doubled fwd/rev target curves; zero-fill pad region [600,640) ---
    for (int e = tid; e < AEXT; e += 128) {
        __bf16 tv = (__bf16)0.0f, rv = (__bf16)0.0f;
        if (e < 2 * K3) {
            int i = e / 3, c = e - 3 * i;
            int im = (i < Nn) ? i : i - Nn;
            tv = (__bf16)tgt[t * K3 + im * 3 + c];
            rv = (__bf16)tgt[t * K3 + (Nn - 1 - im) * 3 + c];
        }
        tgtd[e] = tv;
        revd[e] = rv;
    }
    // --- v2[t] = sum(tgt[t]^2): strided partial + wave reduce + LDS float atomic ---
    {
        float s = 0.0f;
        for (int e = tid; e < K3; e += 128) { float x = tgt[t * K3 + e]; s += x * x; }
        for (int off = 16; off > 0; off >>= 1) s += __shfl_xor(s, off, 32);
        if (lane == 0) atomicAdd(&v2s, s);          // ds_add_f32
    }
    // --- p2 per query ---
    if (tid < 16) {
        float s = 0.0f;
        const float* p = preds + (size_t)(qbase + tid) * K3;
        for (int e = 0; e < K3; ++e) { float x = p[e]; s += x * x; }
        p2s[tid] = s;
    }
    __syncthreads();

    const int   hi   = lane >> 4;
    const float p2   = p2s[lane & 15];
    const float v2   = v2s;
    const float invn = 1.0f / (float)Nn;

    float best = __builtin_inff(); int bestv = 0;
    float open0 = __builtin_inff(), open1 = __builtin_inff();

    // Each wave owns v-tiles wave, wave+4, ... (13 tiles over 4 waves)
    for (int vt = wave; vt < NVT; vt += WAVES) {
        v8f acc = {0.f, 0.f, 0.f, 0.f, 0.f, 0.f, 0.f, 0.f};

        // A-fragment row for this lane (M = lane&15): variant row va.
        // variants[t, va, k] == src[k + 3*(Nn - shift)] on the doubled array.
        int va = vt * 16 + (lane & 15);
        const __bf16* abase; int aoff;
        if (va < Nn)       { abase = tgtd; aoff = 3 * (Nn   - va); }
        else if (va < V2N) { abase = revd; aoff = 3 * (2*Nn - va); }
        else               { abase = tgtd; aoff = 0; }  // pad row; excluded below

        const __bf16* arow = abase + aoff;

        // Double-buffered A fragments: overlap LDS latency with the WMMA pipe.
        v16bf acur, anext;
        #pragma unroll
        for (int i = 0; i < 16; ++i) acur[i] = arow[kmap(i, hi)];   // kt = 0

        #pragma unroll
        for (int kt = 0; kt < NKT; ++kt) {
            if (kt + 1 < NKT) {
                #pragma unroll
                for (int i = 0; i < 16; ++i)
                    anext[i] = arow[(kt + 1) * 32 + kmap(i, hi)];
            }
            v16bf b;
            const __bf16* bptr = &Bfrag[(kt * 32 + lane) * 16];
            #pragma unroll
            for (int i = 0; i < 16; ++i) b[i] = bptr[i];   // re-read from LDS for every vt tile

            acc = __builtin_amdgcn_wmma_f32_16x16x32_bf16(
                false, acur, false, b, (short)0, acc, false, false);
            acur = anext;
        }

        // C layout: VGPR g holds row (g) for lanes 0-15, (g+8) for lanes 16-31
        const int vbase = vt * 16 + (hi ? 8 : 0);
        #pragma unroll
        for (int g = 0; g < 8; ++g) {
            int v = vbase + g;
            float d2 = fmaxf(v2 + p2 - 2.0f * acc[g], 0.0f) * invn;
            bool upd = (v < V2N) && (d2 < best);      // branchless: v_cmp + v_cndmask
            best  = upd ? d2 : best;
            bestv = upd ? v  : bestv;
            open0 = (v == 0)  ? d2 : open0;
            open1 = (v == Nn) ? d2 : open1;
        }
    }

    // Fold lane L with L+16 (same q column) — wave32 shuffles
    {
        float ob = __shfl_xor(best, 16, 32);
        int   ov = __shfl_xor(bestv, 16, 32);
        bool take = (ob < best) || (ob == best && ov < bestv);
        best  = take ? ob : best;
        bestv = take ? ov : bestv;
        open0 = fminf(open0, __shfl_xor(open0, 16, 32));
        open1 = fminf(open1, __shfl_xor(open1, 16, 32));
    }
    if (lane < 16) {
        wmin[wave][lane] = best; widd[wave][lane] = bestv;
        wo0[wave][lane]  = open0; wo1[wave][lane]  = open1;
    }
    __syncthreads();

    // Final cross-wave reduce + class cost + store, one thread per query
    if (tid < 16) {
        float bb = wmin[0][tid]; int bv = widd[0][tid];
        float o0 = wo0[0][tid],  o1 = wo1[0][tid];
        #pragma unroll
        for (int w = 1; w < WAVES; ++w) {
            float m = wmin[w][tid]; int mv = widd[w][tid];
            bool take = (m < bb) || (m == bb && mv < bv);
            bb = take ? m  : bb;
            bv = take ? mv : bv;
            o0 = fminf(o0, wo0[w][tid]);
            o1 = fminf(o1, wo1[w][tid]);
        }
        int icl = iscl[t];
        bool closed = icl > 0;
        float geom = closed ? bb : fminf(o0, o1);
        int   idm  = (bv > Nn - 1) ? (2 * Nn - 1 - bv) : bv;
        int   id   = closed ? idm : 0;
        geom *= wgt[t];

        // cost_class = -log(type_p[lab]+eps) - log(valid_p[0]+eps) - log(closed_p[icl]+eps)
        int q   = qbase + tid;
        int lab = labels[t];
        const float* tp = tlog + q * 4;
        float m4 = fmaxf(fmaxf(tp[0], tp[1]), fmaxf(tp[2], tp[3]));
        float e0 = __expf(tp[0] - m4), e1 = __expf(tp[1] - m4);
        float e2 = __expf(tp[2] - m4), e3 = __expf(tp[3] - m4);
        float s4 = e0 + e1 + e2 + e3;
        float tprob = ((lab == 0) ? e0 : (lab == 1) ? e1 : (lab == 2) ? e2 : e3) / s4;

        const float* vp = vlog + q * 2;
        float mvv = fmaxf(vp[0], vp[1]);
        float ev0 = __expf(vp[0] - mvv), ev1 = __expf(vp[1] - mvv);
        float vprob = ev0 / (ev0 + ev1);

        const float* cp = clog + q * 2;
        float mcc = fmaxf(cp[0], cp[1]);
        float ec0 = __expf(cp[0] - mcc), ec1 = __expf(cp[1] - mcc);
        float cprob = ((icl == 0) ? ec0 : ec1) / (ec0 + ec1);

        float cc = -__logf(tprob + EPSF) - __logf(vprob + EPSF) - __logf(cprob + EPSF);

        outC[(size_t)q * Tn + t] = geom + cc;
        outI[(size_t)q * Tn + t] = (float)id;
    }
}

extern "C" void kernel_launch(void* const* d_in, const int* in_sizes, int n_in,
                              void* d_out, int out_size, void* d_ws, size_t ws_size,
                              hipStream_t stream) {
    (void)in_sizes; (void)n_in; (void)out_size; (void)d_ws; (void)ws_size;
    const float* preds  = (const float*)d_in[0];  // pred_curve_points (1,Q,N,3)
    const float* vlog   = (const float*)d_in[1];  // pred_curve_logits (1,Q,2)
    const float* tlog   = (const float*)d_in[2];  // pred_curve_type   (1,Q,4)
    const float* clog   = (const float*)d_in[3];  // closed_curve_logits (1,Q,2)
    const float* tgt    = (const float*)d_in[4];  // tgt_curve_points (T,N,3)
    const int*   labels = (const int*)d_in[5];    // tgt_labels (T,)
    const int*   iscl   = (const int*)d_in[6];    // tgt_is_closed (T,)
    const float* wgtp   = (const float*)d_in[7];  // curve_length_weighting (T,)

    float* outC = (float*)d_out;
    float* outI = outC + (size_t)Qn * Tn;

    dim3 grid(Qn / 16, Tn);   // 64 x 256 blocks
    matcher_wmma_kernel<<<grid, 128, 0, stream>>>(
        preds, vlog, tlog, clog, tgt, labels, iscl, wgtp, outC, outI);
}
